// Informer_76287209112180
// MI455X (gfx1250) — matrix operand order repaired and checked by reading, not compiled, not run on hardware
//
#include <hip/hip_runtime.h>

typedef __bf16 bf16;
typedef __attribute__((ext_vector_type(16))) __bf16 v16bf;
typedef __attribute__((ext_vector_type(8)))  float  v8f;

#define BB 8
#define LL 2048
#define DM 512
#define DFF 2048
#define NH 8
#define HD 64
#define NSAMP 24
#define NTOP 24

__device__ __forceinline__ unsigned hashu(unsigned x) {
    x ^= x >> 16; x *= 0x7feb352dU;
    x ^= x >> 15; x *= 0x846ca68bU;
    x ^= x >> 16; return x;
}

__device__ __forceinline__ unsigned pack_bf16(float a, float b) {
    union { bf16 h[2]; unsigned u; } p;
    p.h[0] = (bf16)a; p.h[1] = (bf16)b;
    return p.u;
}

// ---------------------------------------------------------------------------
// Input LayerNorm over 32 channels, one thread per (b,l) row.
// ---------------------------------------------------------------------------
__global__ void ln_in_kernel(const float* __restrict__ x, const float* __restrict__ g,
                             const float* __restrict__ bta, float* __restrict__ xn, int rows) {
    int r = blockIdx.x * blockDim.x + threadIdx.x;
    if (r >= rows) return;
    const float* xr = &x[(size_t)r * 32];
    float m = 0.f;
    #pragma unroll
    for (int i = 0; i < 32; ++i) m += xr[i];
    m *= (1.f / 32.f);
    float v = 0.f;
    #pragma unroll
    for (int i = 0; i < 32; ++i) { float d = xr[i] - m; v += d * d; }
    v *= (1.f / 32.f);
    float inv = rsqrtf(v + 1e-5f);
    #pragma unroll
    for (int i = 0; i < 32; ++i)
        xn[(size_t)r * 32 + i] = (xr[i] - m) * inv * g[i] + bta[i];
}

// ---------------------------------------------------------------------------
// Token conv (k=3, wrap) + positional + time embedding. Block per (b,l).
// ---------------------------------------------------------------------------
__global__ void embed_kernel(const float* __restrict__ xn, const float* __restrict__ tf,
                             const float* __restrict__ W_tok, const float* __restrict__ W_time,
                             const float* __restrict__ b_time, float* __restrict__ h) {
    int bl = blockIdx.x;            // 0..B*L-1
    int b = bl >> 11, l = bl & (LL - 1);
    __shared__ float xs[96];
    int tid = threadIdx.x;          // 512 threads
    if (tid < 96) {
        int kk = tid / 32, ci = tid % 32;
        int lsrc = l - 1 + kk;
        if (lsrc < 0) lsrc += LL;
        if (lsrc >= LL) lsrc -= LL;
        xs[tid] = xn[((size_t)(b * LL + lsrc)) * 32 + ci];
    }
    __syncthreads();
    int c = tid;
    float acc = b_time[c];
    int pairI = c >> 1;
    float div = __expf((float)(2 * pairI) * (-9.210340371976184f / (float)DM));
    float ang = (float)l * div;
    acc += (c & 1) ? __cosf(ang) : __sinf(ang);
    const float* tfb = &tf[(size_t)bl * 4];
    #pragma unroll
    for (int j = 0; j < 4; ++j) acc += tfb[j] * W_time[c * 4 + j];
    for (int i = 0; i < 96; ++i) acc += xs[i] * W_tok[(size_t)i * DM + c];
    h[(size_t)bl * DM + c] = acc;
}

// ---------------------------------------------------------------------------
// LayerNorm over 512, block per row (256 threads, 2 elems each). In-place safe.
// ---------------------------------------------------------------------------
__global__ void ln512_kernel(const float* __restrict__ src, const float* __restrict__ g,
                             const float* __restrict__ bta, float* __restrict__ dst) {
    int row = blockIdx.x, tid = threadIdx.x;
    __shared__ float red[256];
    const float* sr = &src[(size_t)row * DM];
    float a0 = sr[tid], a1 = sr[tid + 256];
    red[tid] = a0 + a1; __syncthreads();
    for (int s = 128; s > 0; s >>= 1) { if (tid < s) red[tid] += red[tid + s]; __syncthreads(); }
    float mean = red[0] * (1.f / DM); __syncthreads();
    float d0 = a0 - mean, d1 = a1 - mean;
    red[tid] = d0 * d0 + d1 * d1; __syncthreads();
    for (int s = 128; s > 0; s >>= 1) { if (tid < s) red[tid] += red[tid + s]; __syncthreads(); }
    float inv = rsqrtf(red[0] * (1.f / DM) + 1e-5f);
    float* dr = &dst[(size_t)row * DM];
    dr[tid]       = d0 * inv * g[tid] + bta[tid];
    dr[tid + 256] = d1 * inv * g[tid + 256] + bta[tid + 256];
}

// ---------------------------------------------------------------------------
// Tiled bf16 WMMA GEMM: C[M,N] = A[M,K] @ W[N,K]^T + bias (+residual, relu).
// Block tile 128x128, BK=64, 8 waves (4 over M x 2 over N), wave tile 32x64
// => 16 v_wmma_f32_16x16x32_bf16 per wave per barrier interval.
//
// LDS tiles are stored in FRAGMENT order: [subtile][lane][8 x uint] so each
// lane's 32-byte v16bf operand is one contiguous, 32B-aligned LDS read
// (2 x ds_load_b128, no v_mov_b16 repacking). The swizzle (inverse of the
// ISA 7.12.2 A/B VGPR layouts) is applied during staging. Consecutive
// fragment slots (v0, v0+1), v0 even, map to 4 contiguous K elements in both
// the A and B layouts, so each staging unit is one 16B-aligned float4 global
// load (uint2 for pre-converted bf16 A) -> 2x v_cvt_pk_bf16_f32 -> one
// ds_store_b64. A_BF16 is a template parameter: no branches in the hot loop.
// ---------------------------------------------------------------------------
template <int A_BF16>
__global__ __launch_bounds__(256) void gemm_wmma_kernel(
    const void* __restrict__ Aptr,
    const float* __restrict__ W, const float* __restrict__ bias,
    const float* __restrict__ residual, float* __restrict__ Cf, bf16* __restrict__ Cb,
    int M, int N, int K, int relu) {
    // [msub(8) x kstep(2)][lane][slot] ; [nsub(8) x kstep(2)][lane][slot]
    __shared__ __align__(32) unsigned AsW[16][32][8];
    __shared__ __align__(32) unsigned BsW[16][32][8];
    int tid = threadIdx.x;
    int wave = tid >> 5, lane = tid & 31;
    int laneLo = lane & 15, laneHi = lane >> 4;
    int waveM = wave & 3, waveN = wave >> 2;           // 4 x 2 wave grid
    int row0 = blockIdx.x * 128, col0 = blockIdx.y * 128;
    const float* A32 = (const float*)Aptr;
    const bf16* A16 = (const bf16*)Aptr;

    v8f acc[2][4];
    #pragma unroll
    for (int mi = 0; mi < 2; ++mi)
        #pragma unroll
        for (int ni = 0; ni < 4; ++ni)
            acc[mi][ni] = (v8f){0.f, 0.f, 0.f, 0.f, 0.f, 0.f, 0.f, 0.f};

    for (int k0 = 0; k0 < K; k0 += 64) {
        // Prefetch next weight tile slice (global_prefetch_b8).
        if (k0 + 64 < K)
            __builtin_prefetch(&W[(size_t)(col0 + (tid & 127)) * K + k0 + 64], 0, 1);

        // Fused staging of A and B tiles directly into fragment-order LDS.
        // Unit = 4 elements (slot pair v0, v0+1): float4 load -> ds_store_b64.
        #pragma unroll
        for (int it = 0; it < 8; ++it) {
            int e = tid + it * 256;                 // 0..2047
            int vp = e & 3, ln = (e >> 2) & 31, sub = e >> 7;
            int lo = ln & 15, hi = ln >> 4;
            int msub = sub >> 1, ksp = sub & 1;
            int v0 = vp * 2;                        // 0,2,4,6 (never straddles v=4)
            // ---- A unit: 4 contiguous elements at (row, ka..ka+3)
            int ka = ksp * 32 + 2 * v0 + 8 * hi + (v0 >= 4 ? 8 : 0);
            size_t ga = (size_t)(row0 + msub * 16 + lo) * K + (k0 + ka);
            uint2 pa;
            if (A_BF16) {
                pa = *(const uint2*)&A16[ga];
            } else {
                float4 f = *(const float4*)&A32[ga];
                pa.x = pack_bf16(f.x, f.y);
                pa.y = pack_bf16(f.z, f.w);
            }
            *(uint2*)&AsW[sub][ln][v0] = pa;
            // ---- B unit: 4 contiguous elements at (col, kb..kb+3)
            int kb = ksp * 32 + 2 * v0 + 8 * hi + (v0 >= 4 ? 8 : 0);
            size_t gb = (size_t)(col0 + msub * 16 + lo) * K + (k0 + kb);
            float4 fb = *(const float4*)&W[gb];
            uint2 pb;
            pb.x = pack_bf16(fb.x, fb.y);
            pb.y = pack_bf16(fb.z, fb.w);
            *(uint2*)&BsW[sub][ln][v0] = pb;
        }
        __syncthreads();

        #pragma unroll
        for (int ks = 0; ks < 2; ++ks) {
            v16bf afr[2], bfr[4];
            #pragma unroll
            for (int mi = 0; mi < 2; ++mi)
                afr[mi] = *(const v16bf*)&AsW[(waveM * 2 + mi) * 2 + ks][lane][0];
            #pragma unroll
            for (int ni = 0; ni < 4; ++ni)
                bfr[ni] = *(const v16bf*)&BsW[(waveN * 4 + ni) * 2 + ks][lane][0];
            #pragma unroll
            for (int mi = 0; mi < 2; ++mi)
                #pragma unroll
                for (int ni = 0; ni < 4; ++ni)
                    acc[mi][ni] = __builtin_amdgcn_wmma_f32_16x16x32_bf16(
                        false, afr[mi], false, bfr[ni], (short)0, acc[mi][ni], false, false);
        }
        __syncthreads();
    }

    // Epilogue: C/D layout — VGPR v holds row (v + 8*laneHi), col laneLo.
    #pragma unroll
    for (int mi = 0; mi < 2; ++mi)
        #pragma unroll
        for (int ni = 0; ni < 4; ++ni) {
            #pragma unroll
            for (int vv = 0; vv < 8; ++vv) {
                int gr = row0 + waveM * 32 + mi * 16 + vv + 8 * laneHi;
                int gc = col0 + waveN * 64 + ni * 16 + laneLo;
                float val = acc[mi][ni][vv] + bias[gc];
                if (relu) val = fmaxf(val, 0.f);
                size_t oi = (size_t)gr * N + gc;
                if (residual) val += residual[oi];
                if (Cf) Cf[oi] = val;
                if (Cb) Cb[oi] = (bf16)val;
            }
        }
}

// ---------------------------------------------------------------------------
// ProbSparse attention helpers
// ---------------------------------------------------------------------------
__global__ void meanv_kernel(const float* __restrict__ v, float* __restrict__ meanV) {
    int bh = blockIdx.x, d = threadIdx.x;     // 64 blocks x 64 threads
    int b = bh >> 3, h = bh & 7;
    float s = 0.f;
    for (int l = 0; l < LL; ++l) s += v[((size_t)(b * LL + l)) * DM + h * HD + d];
    meanV[bh * HD + d] = s * (1.f / (float)LL);
}

__global__ void fill_mean_kernel(const float* __restrict__ meanV, float* __restrict__ attn) {
    size_t g = (size_t)blockIdx.x * 256 + threadIdx.x;
    if (g >= (size_t)BB * LL * DM) return;
    int c = (int)(g & (DM - 1));
    int h = c >> 6, d = c & 63;
    size_t bl = g >> 9;
    int b = (int)(bl >> 11);
    attn[g] = meanV[(b * NH + h) * HD + d];
}

__global__ void score_m_kernel(const float* __restrict__ q, const float* __restrict__ k,
                               float* __restrict__ Mbuf, unsigned seed) {
    int g = blockIdx.x * blockDim.x + threadIdx.x;   // B*H*L
    if (g >= BB * NH * LL) return;
    int l = g & (LL - 1), bh = g >> 11;
    int h = bh & 7, b = bh >> 3;
    const float* qp = &q[((size_t)(b * LL + l)) * DM + h * HD];
    float qreg[HD];
    #pragma unroll
    for (int e = 0; e < HD; ++e) qreg[e] = qp[e];
    float mx = -1e30f, sm = 0.f;
    for (int s = 0; s < NSAMP; ++s) {
        unsigned idx = hashu(seed ^ (unsigned)(l * NSAMP + s)) & (LL - 1);
        const float* kp = &k[((size_t)(b * LL + (int)idx)) * DM + h * HD];
        float d = 0.f;
        #pragma unroll
        for (int e = 0; e < HD; ++e) d += qreg[e] * kp[e];
        mx = fmaxf(mx, d); sm += d;
    }
    Mbuf[(size_t)bh * LL + l] = mx - sm * (1.f / (float)LL);
}

__global__ void topk_kernel(const float* __restrict__ Mbuf, int* __restrict__ idxTop) {
    int bh = blockIdx.x, tid = threadIdx.x;   // 256 threads
    __shared__ float mv[LL];
    __shared__ float rv[256];
    __shared__ int ri[256];
    for (int i = tid; i < LL; i += 256) mv[i] = Mbuf[(size_t)bh * LL + i];
    __syncthreads();
    for (int u = 0; u < NTOP; ++u) {
        float bvv = -1e30f; int bii = 0;
        for (int i = tid; i < LL; i += 256) { float x = mv[i]; if (x > bvv) { bvv = x; bii = i; } }
        rv[tid] = bvv; ri[tid] = bii; __syncthreads();
        for (int s = 128; s > 0; s >>= 1) {
            if (tid < s && rv[tid + s] > rv[tid]) { rv[tid] = rv[tid + s]; ri[tid] = ri[tid + s]; }
            __syncthreads();
        }
        if (tid == 0) { idxTop[bh * NTOP + u] = ri[0]; mv[ri[0]] = -1e30f; }
        __syncthreads();
    }
}

__global__ void attn_update_kernel(const float* __restrict__ q, const float* __restrict__ k,
                                   const float* __restrict__ v, const int* __restrict__ idxTop,
                                   float* __restrict__ attn) {
    int blk = blockIdx.x;                 // B*H*NTOP
    int u = blk % NTOP, bh = blk / NTOP;
    int h = bh & 7, b = bh >> 3;
    __shared__ float sc[LL];
    __shared__ float qs[HD];
    __shared__ float red[256];
    __shared__ float part[4][HD];
    int tid = threadIdx.x;                // 256
    int l = idxTop[bh * NTOP + u];
    if (tid < HD) qs[tid] = q[((size_t)(b * LL + l)) * DM + h * HD + tid];
    __syncthreads();
    float lmax = -1e30f;
    for (int kk = tid; kk < LL; kk += 256) {
        const float* kp = &k[((size_t)(b * LL + kk)) * DM + h * HD];
        float d = 0.f;
        #pragma unroll
        for (int e = 0; e < HD; ++e) d += qs[e] * kp[e];
        d *= 0.125f;                      // 1/sqrt(64)
        sc[kk] = d; lmax = fmaxf(lmax, d);
    }
    red[tid] = lmax; __syncthreads();
    for (int s = 128; s > 0; s >>= 1) { if (tid < s) red[tid] = fmaxf(red[tid], red[tid + s]); __syncthreads(); }
    float smax = red[0]; __syncthreads();
    float lsum = 0.f;
    for (int kk = tid; kk < LL; kk += 256) { float e = __expf(sc[kk] - smax); sc[kk] = e; lsum += e; }
    red[tid] = lsum; __syncthreads();
    for (int s = 128; s > 0; s >>= 1) { if (tid < s) red[tid] += red[tid + s]; __syncthreads(); }
    float inv = 1.f / red[0];
    int d = tid & 63, seg = tid >> 6;     // 4 segments of 512 keys
    float acc = 0.f;
    for (int kk = seg * 512; kk < (seg + 1) * 512; ++kk)
        acc += sc[kk] * v[((size_t)(b * LL + kk)) * DM + h * HD + d];
    part[seg][d] = acc; __syncthreads();
    if (tid < HD) {
        float o = (part[0][tid] + part[1][tid] + part[2][tid] + part[3][tid]) * inv;
        attn[((size_t)(b * LL + l)) * DM + h * HD + tid] = o;
    }
}

// ---------------------------------------------------------------------------
// Head MLP: relu(last @ W_pre^T + b_pre) @ W_fc^T + b_fc. Block per batch row.
// ---------------------------------------------------------------------------
__global__ void head_kernel(const float* __restrict__ h, const float* __restrict__ Wp,
                            const float* __restrict__ bp, const float* __restrict__ Wf,
                            const float* __restrict__ bf, float* __restrict__ out) {
    __shared__ float red[256];
    int tid = threadIdx.x, b = blockIdx.x;
    const float* last = &h[((size_t)(b * LL + (LL - 1))) * DM];
    float acc = bp[tid];
    for (int i = 0; i < DM; ++i) acc += last[i] * Wp[(size_t)tid * DM + i];
    float pre = fmaxf(acc, 0.f);
    red[tid] = pre * Wf[tid]; __syncthreads();
    for (int s = 128; s > 0; s >>= 1) { if (tid < s) red[tid] += red[tid + s]; __syncthreads(); }
    if (tid == 0) out[b] = red[0] + bf[0];
}

// ---------------------------------------------------------------------------
extern "C" void kernel_launch(void* const* d_in, const int* in_sizes, int n_in,
                              void* d_out, int out_size, void* d_ws, size_t ws_size,
                              hipStream_t stream) {
    const float* x      = (const float*)d_in[0];
    const float* tf     = (const float*)d_in[1];
    const float* g_in   = (const float*)d_in[2];
    const float* b_in   = (const float*)d_in[3];
    const float* W_tok  = (const float*)d_in[4];
    const float* W_time = (const float*)d_in[5];
    const float* b_time = (const float*)d_in[6];
    const float* Wq     = (const float*)d_in[7];
    const float* bq     = (const float*)d_in[8];
    const float* Wk     = (const float*)d_in[9];
    const float* bk     = (const float*)d_in[10];
    const float* Wv     = (const float*)d_in[11];
    const float* bv     = (const float*)d_in[12];
    const float* Wo     = (const float*)d_in[13];
    const float* bo     = (const float*)d_in[14];
    const float* W1     = (const float*)d_in[15];
    const float* b1     = (const float*)d_in[16];
    const float* W2     = (const float*)d_in[17];
    const float* b2     = (const float*)d_in[18];
    const float* g1     = (const float*)d_in[19];
    const float* be1    = (const float*)d_in[20];
    const float* g2     = (const float*)d_in[21];
    const float* be2    = (const float*)d_in[22];
    const float* g_enc  = (const float*)d_in[23];
    const float* b_enc  = (const float*)d_in[24];
    const float* W_pre  = (const float*)d_in[25];
    const float* b_pre  = (const float*)d_in[26];
    const float* W_fc   = (const float*)d_in[27];
    const float* b_fc   = (const float*)d_in[28];
    (void)in_sizes; (void)n_in; (void)out_size; (void)ws_size;

    const size_t M = (size_t)BB * LL;       // 16384
    size_t off = 0;
    auto alloc = [&](size_t bytes) -> void* {
        void* p = (char*)d_ws + off;
        off += (bytes + 255) & ~(size_t)255;
        return p;
    };
    float* xn    = (float*)alloc(M * 32 * 4);
    float* h     = (float*)alloc(M * DM * 4);
    float* qb    = (float*)alloc(M * DM * 4);
    float* kb    = (float*)alloc(M * DM * 4);
    float* vb    = (float*)alloc(M * DM * 4);
    float* attn  = (float*)alloc(M * DM * 4);
    bf16*  ffnb  = (bf16*)alloc(M * DFF * 2);
    float* ybuf  = (float*)alloc(M * DM * 4);
    float* Mbuf  = (float*)alloc((size_t)BB * NH * LL * 4);
    int*   idxT  = (int*)alloc((size_t)BB * NH * NTOP * 4);
    float* meanV = (float*)alloc((size_t)BB * NH * HD * 4);

    ln_in_kernel<<<(int)((M + 255) / 256), 256, 0, stream>>>(x, g_in, b_in, xn, (int)M);
    embed_kernel<<<(int)M, DM, 0, stream>>>(xn, tf, W_tok, W_time, b_time, h);

    dim3 g512((int)(M / 128), DM / 128);    // N=512
    dim3 g2048((int)(M / 128), DFF / 128);  // N=2048
    size_t fillN = ((size_t)BB * LL * DM + 255) / 256;

    for (int e = 0; e < 2; ++e) {
        const float* Wq_e = Wq + (size_t)e * DM * DM;  const float* bq_e = bq + (size_t)e * DM;
        const float* Wk_e = Wk + (size_t)e * DM * DM;  const float* bk_e = bk + (size_t)e * DM;
        const float* Wv_e = Wv + (size_t)e * DM * DM;  const float* bv_e = bv + (size_t)e * DM;
        const float* Wo_e = Wo + (size_t)e * DM * DM;  const float* bo_e = bo + (size_t)e * DM;
        const float* W1_e = W1 + (size_t)e * DFF * DM; const float* b1_e = b1 + (size_t)e * DFF;
        const float* W2_e = W2 + (size_t)e * DM * DFF; const float* b2_e = b2 + (size_t)e * DM;
        const float* g1_e = g1 + (size_t)e * DM;  const float* be1_e = be1 + (size_t)e * DM;
        const float* g2_e = g2 + (size_t)e * DM;  const float* be2_e = be2 + (size_t)e * DM;

        gemm_wmma_kernel<0><<<g512, 256, 0, stream>>>(h, Wq_e, bq_e, nullptr, qb, nullptr, (int)M, DM, DM, 0);
        gemm_wmma_kernel<0><<<g512, 256, 0, stream>>>(h, Wk_e, bk_e, nullptr, kb, nullptr, (int)M, DM, DM, 0);
        gemm_wmma_kernel<0><<<g512, 256, 0, stream>>>(h, Wv_e, bv_e, nullptr, vb, nullptr, (int)M, DM, DM, 0);

        meanv_kernel<<<BB * NH, HD, 0, stream>>>(vb, meanV);
        fill_mean_kernel<<<(int)fillN, 256, 0, stream>>>(meanV, attn);
        score_m_kernel<<<(BB * NH * LL) / 256, 256, 0, stream>>>(qb, kb, Mbuf, 0x9E3779B9u * (unsigned)(e + 1));
        topk_kernel<<<BB * NH, 256, 0, stream>>>(Mbuf, idxT);
        attn_update_kernel<<<BB * NH * NTOP, 256, 0, stream>>>(qb, kb, vb, idxT, attn);

        gemm_wmma_kernel<0><<<g512, 256, 0, stream>>>(attn, Wo_e, bo_e, h, ybuf, nullptr, (int)M, DM, DM, 0);
        ln512_kernel<<<(int)M, 256, 0, stream>>>(ybuf, g1_e, be1_e, h);

        gemm_wmma_kernel<0><<<g2048, 256, 0, stream>>>(h, W1_e, b1_e, nullptr, nullptr, ffnb, (int)M, DFF, DM, 1);
        gemm_wmma_kernel<1><<<g512, 256, 0, stream>>>(ffnb, W2_e, b2_e, h, ybuf, nullptr, (int)M, DM, DFF, 0);
        ln512_kernel<<<(int)M, 256, 0, stream>>>(ybuf, g2_e, be2_e, h);
    }

    ln512_kernel<<<(int)M, 256, 0, stream>>>(h, g_enc, b_enc, h);
    head_kernel<<<BB, 256, 0, stream>>>(h, W_pre, b_pre, W_fc, b_fc, (float*)d_out);
}
